// CrossAttentionB_35270271435503
// MI455X (gfx1250) — matrix operand order repaired and checked by reading, not compiled, not run on hardware
//
#include <hip/hip_runtime.h>
#include <hip/hip_bf16.h>

// B=8, SQ=2048, SKV=2048, D_IN=1024, D_EMB=512, D_OUT=512
#define B_    8
#define SQ_   2048
#define SKV_  2048
#define DIN_  1024
#define DEMB_ 512
#define DOUT_ 512

#if defined(__HIP_DEVICE_COMPILE__) && defined(__gfx1250__) && \
    __has_builtin(__builtin_amdgcn_global_load_async_to_lds_b128) && \
    __has_builtin(__builtin_amdgcn_s_wait_asynccnt)
#define USE_ASYNC_LDS 1
#else
#define USE_ASYNC_LDS 0
#endif

typedef __attribute__((ext_vector_type(16))) __bf16 v16bf;
typedef __attribute__((ext_vector_type(8)))  __bf16 v8bf;
typedef __attribute__((ext_vector_type(8)))  float  v8f;
typedef __attribute__((ext_vector_type(4)))  int    v4i;

#if USE_ASYNC_LDS
typedef __attribute__((address_space(1))) v4i gv4i;   // global int4
typedef __attribute__((address_space(3))) v4i lv4i;   // LDS int4
#endif

__device__ __forceinline__ v8f zero8() {
  v8f z;
#pragma unroll
  for (int i = 0; i < 8; ++i) z[i] = 0.0f;
  return z;
}

// A fragment (16x32 bf16) from row-major [m][k] LDS image.
// lane: m=lane&15, half=lane>>4; element e -> K = (e&7) + 8*half + 16*(e>>3)
__device__ __forceinline__ v16bf frag_a_ld(const __bf16* base, int row0, int k0, int ld) {
  const int lane = threadIdx.x & 31;
  const int m = lane & 15, h = lane >> 4;
  const __bf16* p = base + (size_t)(row0 + m) * ld + k0 + 8 * h;
  v8bf lo = *(const v8bf*)(p);
  v8bf hi = *(const v8bf*)(p + 16);
  return __builtin_shufflevector(lo, hi, 0, 1, 2, 3, 4, 5, 6, 7, 8, 9, 10, 11, 12, 13, 14, 15);
}

// B fragment (32x16 bf16) from a row-major [n][k] LDS image (B^T stored row-major).
// lane: n=lane&15, half=lane>>4; element e -> K = (e&7) + 8*half + 16*(e>>3)
__device__ __forceinline__ v16bf frag_b_ld(const __bf16* base, int n0, int k0, int ld) {
  const int lane = threadIdx.x & 31;
  const int n = lane & 15, h = lane >> 4;
  const __bf16* p = base + (size_t)(n0 + n) * ld + k0 + 8 * h;
  v8bf lo = *(const v8bf*)(p);
  v8bf hi = *(const v8bf*)(p + 16);
  return __builtin_shufflevector(lo, hi, 0, 1, 2, 3, 4, 5, 6, 7, 8, 9, 10, 11, 12, 13, 14, 15);
}

__device__ __forceinline__ v8f wmma_bf16(v16bf a, v16bf b, v8f c) {
  return __builtin_amdgcn_wmma_f32_16x16x32_bf16(false, a, false, b, (short)0, c, false, false);
}

// ---------------------------------------------------------------------------
// 64x512 bf16 tile copy, global -> LDS (dst ld = 520), 256 threads.
// Async path: GLOBAL_LOAD_ASYNC_TO_LDS_B128 (ASYNCcnt), no VGPR staging.
// Fallback: fully unrolled, batched loads-then-stores (keeps 8 loads in flight).
// ---------------------------------------------------------------------------
__device__ __forceinline__ void tile_copy_issue(__bf16* dst, const __bf16* src) {
  const int tid = threadIdx.x;
#if USE_ASYNC_LDS
#pragma unroll
  for (int it = 0; it < 16; ++it) {
    int e = (tid + it * 256) * 8;
    int r = e >> 9, cc = e & 511;
    __builtin_amdgcn_global_load_async_to_lds_b128(
        (gv4i*)(src + (size_t)r * DEMB_ + cc),
        (lv4i*)(dst + r * 520 + cc), 0, 0);
  }
#else
#pragma unroll
  for (int half = 0; half < 2; ++half) {
    v8bf tmp[8];
#pragma unroll
    for (int it = 0; it < 8; ++it) {
      int e = (tid + (half * 8 + it) * 256) * 8;
      int r = e >> 9, cc = e & 511;
      tmp[it] = *(const v8bf*)(src + (size_t)r * DEMB_ + cc);
    }
#pragma unroll
    for (int it = 0; it < 8; ++it) {
      int e = (tid + (half * 8 + it) * 256) * 8;
      int r = e >> 9, cc = e & 511;
      *(v8bf*)(dst + r * 520 + cc) = tmp[it];
    }
  }
#endif
}

__device__ __forceinline__ void tile_copy_wait() {
#if USE_ASYNC_LDS
  __builtin_amdgcn_s_wait_asynccnt(0);
#endif
}

// ---------------------------------------------------------------------------
// fp32 -> bf16 elementwise convert (en_out -> K/V workspace)
// ---------------------------------------------------------------------------
__global__ void cvt_bf16_kernel(const float* __restrict__ in, __bf16* __restrict__ out, int n) {
  int i = blockIdx.x * blockDim.x + threadIdx.x;
  int stride = gridDim.x * blockDim.x;
  for (; i < n; i += stride) out[i] = (__bf16)in[i];
}

// ---------------------------------------------------------------------------
// C[M,N] = A[M,K] * W[K,N] + bias  (bf16 WMMA, fp32 accumulate)
// 128 threads = 4 waves. Tile: 64(M) x 64(N), K stepped by 32.
// ---------------------------------------------------------------------------
__global__ __launch_bounds__(128) void gemm_bias_kernel(
    const float* __restrict__ Af32, const __bf16* __restrict__ Abf16,
    const float* __restrict__ W, const float* __restrict__ bias,
    float* __restrict__ outF, __bf16* __restrict__ outB, int K, int N) {
  __shared__ __bf16 At[64 * 40];   // [m][k], ld=40 (bank-conflict pad)
  __shared__ __bf16 Bt[64 * 40];   // [n][k], ld=40

  const int tid = threadIdx.x;
  const int w = tid >> 5;
  const int lane = tid & 31;
  const int h = lane >> 4;
  const int m0 = blockIdx.x * 64;
  const int n0 = blockIdx.y * 64;

  v8f c[4];
#pragma unroll
  for (int j = 0; j < 4; ++j) c[j] = zero8();

  for (int k0 = 0; k0 < K; k0 += 32) {
    __syncthreads();
    // --- A tile 64x32 -> LDS bf16 (batched) ---
    if (Abf16) {
#pragma unroll
      for (int it = 0; it < 2; ++it) {
        int e = (tid + it * 128) * 8;
        int m = e >> 5, k = e & 31;
        *(v8bf*)(At + m * 40 + k) = *(const v8bf*)(Abf16 + (size_t)(m0 + m) * K + k0 + k);
      }
    } else {
      float4 f[4];
#pragma unroll
      for (int it = 0; it < 4; ++it) {
        int e = (tid + it * 128) * 4;
        int m = e >> 5, k = e & 31;
        f[it] = *(const float4*)(Af32 + (size_t)(m0 + m) * K + k0 + k);
      }
#pragma unroll
      for (int it = 0; it < 4; ++it) {
        int e = (tid + it * 128) * 4;
        int m = e >> 5, k = e & 31;
        __bf16* d2 = At + m * 40 + k;
        d2[0] = (__bf16)f[it].x; d2[1] = (__bf16)f[it].y;
        d2[2] = (__bf16)f[it].z; d2[3] = (__bf16)f[it].w;
      }
    }
    // --- W tile 32x64 -> [n][k] bf16 (batched float4 reads, transposed stores) ---
    {
      float4 g[4];
#pragma unroll
      for (int it = 0; it < 4; ++it) {
        int i = tid + it * 128;            // [0,512)
        int k = i >> 4, n4 = (i & 15) * 4;
        g[it] = *(const float4*)(W + (size_t)(k0 + k) * N + n0 + n4);
      }
#pragma unroll
      for (int it = 0; it < 4; ++it) {
        int i = tid + it * 128;
        int k = i >> 4, n4 = (i & 15) * 4;
        Bt[(n4 + 0) * 40 + k] = (__bf16)g[it].x;
        Bt[(n4 + 1) * 40 + k] = (__bf16)g[it].y;
        Bt[(n4 + 2) * 40 + k] = (__bf16)g[it].z;
        Bt[(n4 + 3) * 40 + k] = (__bf16)g[it].w;
      }
    }
    __syncthreads();

    v16bf a = frag_a_ld(At, 16 * w, 0, 40);
#pragma unroll
    for (int j = 0; j < 4; ++j) {
      c[j] = wmma_bf16(a, frag_b_ld(Bt, 16 * j, 0, 40), c[j]);
    }
  }

#pragma unroll
  for (int j = 0; j < 4; ++j) {
#pragma unroll
    for (int r = 0; r < 8; ++r) {
      int row = m0 + 16 * w + r + 8 * h;
      int col = n0 + 16 * j + (lane & 15);
      float v = c[j][r] + bias[col];
      size_t idx = (size_t)row * N + col;
      if (outF) outF[idx] = v;
      if (outB) outB[idx] = (__bf16)v;
    }
  }
}

// ---------------------------------------------------------------------------
// Fused attention: workgroup = (batch b, 64 query rows). 256 threads =
// 8 waves (wm 0..3) x (wn 0..1). Double-buffered K/V chunks (async-to-LDS).
// Pass 1: online row max / sum-exp of S = scale * q K^T.
// Pass 2: recompute S, write normalized P (fp32) to attention output, stage
//         P bf16 + V^T in LDS, accumulate O += P V with WMMA.
// Epilogue: sum_out = O + q (fp32 residual), stored bf16 for proj GEMM.
// Dynamic LDS: q(64x520) en0 en1 vt(512x72) p(64x72) stats = 284,160 B.
// ---------------------------------------------------------------------------
__global__ __launch_bounds__(256) void attn_kernel(
    const __bf16* __restrict__ qb, const float* __restrict__ qf,
    const __bf16* __restrict__ kv, float* __restrict__ attn,
    __bf16* __restrict__ sum_out) {
  extern __shared__ __align__(16) char smem[];
  __bf16* q_lds   = (__bf16*)(smem);             // [64][520]
  __bf16* en0_lds = (__bf16*)(smem + 66560);     // [64][520]
  __bf16* en1_lds = (__bf16*)(smem + 133120);    // [64][520]
  __bf16* vt_lds  = (__bf16*)(smem + 199680);    // [512][72]
  __bf16* p_lds   = (__bf16*)(smem + 273408);    // [64][72]
  float*  m_run   = (float*)(smem + 282624);     // [64]
  float*  l_run   = m_run + 64;                  // [64]
  float*  mx2     = l_run + 64;                  // [2][64]
  float*  ls2     = mx2 + 128;                   // [2][64]
  __bf16* enb[2]  = {en0_lds, en1_lds};

  const int tid = threadIdx.x;
  const int lane = tid & 31;
  const int w = tid >> 5;
  const int wm = w & 3;       // M tile (16 rows)
  const int wn = w >> 2;      // N half
  const int h = lane >> 4;
  const int b = blockIdx.x >> 5;      // 32 q-blocks per batch
  const int qblk = blockIdx.x & 31;
  const int qrow0 = qblk * 64;
  const float scale = 0.044194173824159216f;   // 1/sqrt(512)

  const __bf16* qsrc = qb + ((size_t)b * SQ_ + qrow0) * DEMB_;
  const __bf16* kvb  = kv + (size_t)b * SKV_ * DEMB_;

  // q tile (64 x 512 bf16) + stats init, then preload K/V chunk 0
  tile_copy_issue(q_lds, qsrc);
  if (tid < 64) { m_run[tid] = -3.0e38f; l_run[tid] = 0.0f; }
  tile_copy_wait();
  tile_copy_issue(enb[0], kvb);
  tile_copy_wait();
  __syncthreads();

  // ---------------- pass 1: softmax statistics ----------------
  for (int t0 = 0; t0 < SKV_; t0 += 64) {
    const int cur = (t0 >> 6) & 1;
    const __bf16* en_lds = enb[cur];
    if (t0 + 64 < SKV_) tile_copy_issue(enb[cur ^ 1], kvb + (size_t)(t0 + 64) * DEMB_);

    v8f s0 = zero8(), s1 = zero8();
#pragma unroll
    for (int k0 = 0; k0 < DEMB_; k0 += 32) {
      v16bf a = frag_a_ld(q_lds, 16 * wm, k0, 520);
      s0 = wmma_bf16(a, frag_b_ld(en_lds, 32 * wn, k0, 520), s0);
      s1 = wmma_bf16(a, frag_b_ld(en_lds, 32 * wn + 16, k0, 520), s1);
    }

#pragma unroll
    for (int r = 0; r < 8; ++r) {
      float a0 = s0[r] * scale, a1 = s1[r] * scale;
      float mv = fmaxf(a0, a1);
#pragma unroll
      for (int off = 8; off >= 1; off >>= 1) mv = fmaxf(mv, __shfl_xor(mv, off, 32));
      float sv = __expf(a0 - mv) + __expf(a1 - mv);
#pragma unroll
      for (int off = 8; off >= 1; off >>= 1) sv += __shfl_xor(sv, off, 32);
      if ((lane & 15) == 0) {
        int row = 16 * wm + r + 8 * h;
        mx2[wn * 64 + row] = mv;
        ls2[wn * 64 + row] = sv;
      }
    }
    __syncthreads();
    if (tid < 64) {
      float m0v = mx2[tid], m1v = mx2[64 + tid];
      float mc = fmaxf(m0v, m1v);
      float lc = ls2[tid] * __expf(m0v - mc) + ls2[64 + tid] * __expf(m1v - mc);
      float mo = m_run[tid], lo = l_run[tid];
      float mn = fmaxf(mo, mc);
      l_run[tid] = lo * __expf(mo - mn) + lc * __expf(mc - mn);
      m_run[tid] = mn;
    }
    tile_copy_wait();
    __syncthreads();
  }

  // ---------------- pass 2: P out + O = P V ----------------
  v8f o[16];
#pragma unroll
  for (int j = 0; j < 16; ++j) o[j] = zero8();

  tile_copy_issue(enb[0], kvb);
  tile_copy_wait();
  __syncthreads();

  for (int t0 = 0; t0 < SKV_; t0 += 64) {
    const int cur = (t0 >> 6) & 1;
    const __bf16* en_lds = enb[cur];
    if (t0 + 64 < SKV_) tile_copy_issue(enb[cur ^ 1], kvb + (size_t)(t0 + 64) * DEMB_);

    // V^T staging: vt[d][t] = en[t][d]  (b128 row reads, scalar strided writes)
#pragma unroll 4
    for (int it = 0; it < 16; ++it) {
      int i = tid + it * 256;            // [0,4096)
      int tt = i & 63, d8 = i >> 6;
      v8bf vv = *(const v8bf*)(en_lds + tt * 520 + d8 * 8);
#pragma unroll
      for (int j = 0; j < 8; ++j) vt_lds[(d8 * 8 + j) * 72 + tt] = vv[j];
    }

    v8f s0 = zero8(), s1 = zero8();
#pragma unroll
    for (int k0 = 0; k0 < DEMB_; k0 += 32) {
      v16bf a = frag_a_ld(q_lds, 16 * wm, k0, 520);
      s0 = wmma_bf16(a, frag_b_ld(en_lds, 32 * wn, k0, 520), s0);
      s1 = wmma_bf16(a, frag_b_ld(en_lds, 32 * wn + 16, k0, 520), s1);
    }

#pragma unroll
    for (int r = 0; r < 8; ++r) {
      int row = 16 * wm + r + 8 * h;
      float mv = m_run[row];
      float li = 1.0f / l_run[row];
      float p0 = __expf(s0[r] * scale - mv) * li;
      float p1 = __expf(s1[r] * scale - mv) * li;
      size_t ai = ((size_t)b * SQ_ + qrow0 + row) * (size_t)SKV_ + t0 + 32 * wn + (lane & 15);
      attn[ai] = p0;
      attn[ai + 16] = p1;
      p_lds[row * 72 + 32 * wn + (lane & 15)] = (__bf16)p0;
      p_lds[row * 72 + 32 * wn + 16 + (lane & 15)] = (__bf16)p1;
    }
    __syncthreads();  // P + V^T ready

#pragma unroll
    for (int k0 = 0; k0 < 64; k0 += 32) {
      v16bf a = frag_a_ld(p_lds, 16 * wm, k0, 72);
#pragma unroll
      for (int j = 0; j < 16; ++j) {
        o[j] = wmma_bf16(a, frag_b_ld(vt_lds, 256 * wn + 16 * j, k0, 72), o[j]);
      }
    }
    tile_copy_wait();
    __syncthreads();  // protect vt/p overwrite; next en buffer ready
  }

  // epilogue: residual + bf16 store of sum_out
  const float* qfp = qf + ((size_t)b * SQ_ + qrow0) * DEMB_;
  __bf16* sout = sum_out + ((size_t)b * SQ_ + qrow0) * DEMB_;
#pragma unroll
  for (int j = 0; j < 16; ++j) {
#pragma unroll
    for (int r = 0; r < 8; ++r) {
      int row = 16 * wm + r + 8 * h;
      int col = 256 * wn + 16 * j + (lane & 15);
      float v = o[j][r] + qfp[(size_t)row * DEMB_ + col];
      sout[(size_t)row * DEMB_ + col] = (__bf16)v;
    }
  }
}

// ---------------------------------------------------------------------------
extern "C" void kernel_launch(void* const* d_in, const int* in_sizes, int n_in,
                              void* d_out, int out_size, void* d_ws, size_t ws_size,
                              hipStream_t stream) {
  (void)in_sizes; (void)n_in; (void)out_size; (void)ws_size;
  const float* x1 = (const float*)d_in[0];   // [8,2048,1024]
  const float* en = (const float*)d_in[1];   // [8,2048,512]
  const float* Wq = (const float*)d_in[2];   // [1024,512]
  const float* bq = (const float*)d_in[3];   // [512]
  const float* Wf = (const float*)d_in[4];   // [512,512]
  const float* bf = (const float*)d_in[5];   // [512]

  float* proj = (float*)d_out;                               // [8,2048,512]
  float* attn = proj + (size_t)B_ * SQ_ * DOUT_;             // [8,2048,2048]

  char* ws = (char*)d_ws;
  float*  ws_qf  = (float*)(ws);                 // q fp32   : 33,554,432 B
  __bf16* ws_qb  = (__bf16*)(ws + 33554432);     // q bf16   : 16,777,216 B
  __bf16* ws_kv  = (__bf16*)(ws + 50331648);     // en bf16  : 16,777,216 B
  __bf16* ws_sum = (__bf16*)(ws + 67108864);     // sum bf16 : 16,777,216 B

  // 1) K/V -> bf16 (L2-resident thereafter: 16 MB << 192 MB L2)
  cvt_bf16_kernel<<<4096, 256, 0, stream>>>(en, ws_kv, B_ * SKV_ * DEMB_);

  // 2) q = x1 @ Wq + bq  (fp32 + bf16 copies)
  gemm_bias_kernel<<<dim3((B_ * SQ_) / 64, DEMB_ / 64), 128, 0, stream>>>(
      x1, nullptr, Wq, bq, ws_qf, ws_qb, DIN_, DEMB_);

  // 3) fused attention (writes attention to d_out, sum_out bf16 to ws)
  attn_kernel<<<B_ * (SQ_ / 64), 256, 284160, stream>>>(ws_qb, ws_qf, ws_kv, attn, ws_sum);

  // 4) proj = sum_out @ Wf + bf
  gemm_bias_kernel<<<dim3((B_ * SQ_) / 64, DOUT_ / 64), 128, 0, stream>>>(
      nullptr, ws_sum, Wf, bf, proj, nullptr, DEMB_, DOUT_);
}
